// CausalSelfAttentionRelativePosition_87385404604771
// MI455X (gfx1250) — matrix operand order repaired and checked by reading, not compiled, not run on hardware
//
#include <hip/hip_runtime.h>
#include <hip/hip_bf16.h>
#include <stdint.h>

// CDNA5 / gfx1250 causal self-attention with relative position (Music Transformer).
// bf16 WMMA (v_wmma_f32_16x16x32_bf16) for all GEMM-like work, f32 accumulate.
// GEMM LDS staging via Tensor Data Mover (tensor_load_to_lds + TENSORcnt), double-buffered.

typedef __bf16 bf16;
typedef __attribute__((ext_vector_type(16))) __bf16 bf16x16;
typedef __attribute__((ext_vector_type(8)))  __bf16 bf16x8;
typedef __attribute__((ext_vector_type(8)))  float  f32x8;
typedef __attribute__((ext_vector_type(4)))  uint32_t u32x4;
typedef __attribute__((ext_vector_type(8)))  uint32_t u32x8;

#define B_DIM  2
#define T_DIM  2048
#define H_DIM  16
#define HS_DIM 64
#define C_DIM  1024

__device__ __forceinline__ bf16x16 cat8(bf16x8 lo, bf16x8 hi) {
  return __builtin_shufflevector(lo, hi, 0,1,2,3,4,5,6,7,8,9,10,11,12,13,14,15);
}

// A-matrix 16x32 fragment from row-major [16 x >=32] storage.
// lane: M = lane&15, half = lane>>4; elems e -> K = half*16+e (contiguous 16)
__device__ __forceinline__ bf16x16 load_frag_a(const bf16* p, int stride, int lane) {
  const bf16* r = p + (lane & 15) * stride + ((lane >> 4) << 4);
  bf16x8 lo = *(const bf16x8*)r;
  bf16x8 hi = *(const bf16x8*)(r + 8);
  return cat8(lo, hi);
}

// B-matrix 32x16 fragment from TRANSPOSED (N x K row-major) storage.
// lane: N = lane&15, half = lane>>4; elems e -> K = half*16+e (contiguous 16)
__device__ __forceinline__ bf16x16 load_frag_bt(const bf16* p, int stride, int lane) {
  const bf16* r = p + (lane & 15) * stride + ((lane >> 4) << 4);
  bf16x8 lo = *(const bf16x8*)r;
  bf16x8 hi = *(const bf16x8*)(r + 8);
  return cat8(lo, hi);
}

__device__ __forceinline__ f32x8 wmma_bf16(bf16x16 a, bf16x16 b, f32x8 c) {
  return __builtin_amdgcn_wmma_f32_16x16x32_bf16(false, a, false, b, (short)0, c, false, false);
}

// ---- Tensor Data Mover: 2D tile (elems_x x rows) of 2-byte elements, row stride in elems.
// D# per CDNA5 ISA ch.8: group0 {count, lds_addr, global_addr, type=2},
// group1 {data_size=1(2B), tensor_dim0/1 = tile dims, tile_dim0/1, dim0 stride}.
__device__ __forceinline__ void tdm_load_2d(unsigned lds_byte, const bf16* gptr,
                                            unsigned elems_x, unsigned rows,
                                            unsigned stride_elems) {
  unsigned long long ga = (unsigned long long)(uintptr_t)gptr;
  u32x4 g0;
  g0[0] = 1u;                                            // count=1, user mode
  g0[1] = lds_byte;                                      // lds_addr[31:0]
  g0[2] = (uint32_t)(ga & 0xffffffffull);                // global_addr[31:0]
  g0[3] = (uint32_t)((ga >> 32) & 0x1ffffffull) | (2u << 30);  // addr[56:32] | type=2
  u32x8 g1;
  g1[0] = 1u << 16;                                      // data_size=1 (2 bytes)
  g1[1] = (elems_x & 0xffffu) << 16;                     // tensor_dim0[15:0]
  g1[2] = ((elems_x >> 16) & 0xffffu) | ((rows & 0xffffu) << 16);  // dim0 hi | tensor_dim1 lo
  g1[3] = ((rows >> 16) & 0xffffu) | ((elems_x & 0xffffu) << 16);  // dim1 hi | tile_dim0
  g1[4] = rows & 0xffffu;                                // tile_dim1 (tile_dim2=0)
  g1[5] = stride_elems;                                  // tensor_dim0_stride[31:0]
  g1[6] = 0u;                                            // stride hi | dim1_stride lo
  g1[7] = 0u;
  u32x4 gz = {0u, 0u, 0u, 0u};
  asm volatile("tensor_load_to_lds %0, %1, %2, %3"
               :: "s"(g0), "s"(g1), "s"(gz), "s"(gz)
               : "memory");
}

// ---------------- convert kernels ----------------
__global__ void cvt_f32_bf16(const float* __restrict__ in, bf16* __restrict__ out, int n) {
  for (int i = blockIdx.x * blockDim.x + threadIdx.x; i < n; i += gridDim.x * blockDim.x)
    out[i] = (bf16)in[i];
}

// in: K x N (row-major f32), out: N x K (row-major bf16)
__global__ void transpose_cvt(const float* __restrict__ in, bf16* __restrict__ out, int K, int N) {
  int total = K * N;
  for (int i = blockIdx.x * blockDim.x + threadIdx.x; i < total; i += gridDim.x * blockDim.x) {
    int n = i / K, k = i - n * K;
    out[i] = (bf16)in[k * N + n];
  }
}

// ---------------- WMMA GEMM: C[M,N] = A[M,K] * BT[N,K]^T ----------------
// LDS tiles staged by the Tensor Data Mover (wave 0 issues descriptors), double-buffered.
// MODE 0: write f32 row-major to Cf.   MODE 1: scatter qkv into Q, K, Vt (bf16).
template <int MODE>
__global__ __launch_bounds__(256) void gemm_wmma(const bf16* __restrict__ A,
                                                 const bf16* __restrict__ BT,
                                                 float* __restrict__ Cf,
                                                 bf16* __restrict__ Qb,
                                                 bf16* __restrict__ Kb,
                                                 bf16* __restrict__ Vtb,
                                                 int M, int N, int K) {
  __shared__ __align__(16) bf16 sA[2][128 * 32];
  __shared__ __align__(16) bf16 sB[2][128 * 32];
  const int tid = threadIdx.x, lane = tid & 31, w = tid >> 5;
  const int m_base = blockIdx.x * 128, n_base = blockIdx.y * 128;
  const int m_off = (w & 1) * 64, n_off = (w >> 1) * 32;

  f32x8 acc[4][2] = {};

  // prologue: TDM-load first K-slice into buffer 0 (one wave issues; EXEC ignored by TDM)
  if (w == 0) {
    tdm_load_2d((unsigned)(uintptr_t)&sA[0][0], A + (size_t)m_base * K, 32u, 128u, (unsigned)K);
    tdm_load_2d((unsigned)(uintptr_t)&sB[0][0], BT + (size_t)n_base * K, 32u, 128u, (unsigned)K);
  }

  for (int kk = 0; kk < K; kk += 32) {
    const int p = (kk >> 5) & 1;
    if (w == 0) __builtin_amdgcn_s_wait_tensorcnt(0);  // wave 0's DMA landed
    __syncthreads();                                   // publish tile to all waves
    if (w == 0 && kk + 32 < K) {                       // prefetch next slice into other buffer
      tdm_load_2d((unsigned)(uintptr_t)&sA[p ^ 1][0],
                  A + (size_t)m_base * K + kk + 32, 32u, 128u, (unsigned)K);
      tdm_load_2d((unsigned)(uintptr_t)&sB[p ^ 1][0],
                  BT + (size_t)n_base * K + kk + 32, 32u, 128u, (unsigned)K);
    }

    bf16x16 af[4], bfr[2];
#pragma unroll
    for (int i = 0; i < 4; ++i) af[i] = load_frag_a(&sA[p][(m_off + i * 16) * 32], 32, lane);
#pragma unroll
    for (int j = 0; j < 2; ++j) bfr[j] = load_frag_bt(&sB[p][(n_off + j * 16) * 32], 32, lane);
#pragma unroll
    for (int i = 0; i < 4; ++i)
#pragma unroll
      for (int j = 0; j < 2; ++j)
        acc[i][j] = wmma_bf16(af[i], bfr[j], acc[i][j]);
  }

  const int half = lane >> 4, cl = lane & 15;
#pragma unroll
  for (int i = 0; i < 4; ++i)
#pragma unroll
    for (int j = 0; j < 2; ++j)
#pragma unroll
      for (int r = 0; r < 8; ++r) {
        int row = m_base + m_off + i * 16 + r + half * 8;
        int col = n_base + n_off + j * 16 + cl;
        float v = acc[i][j][r];
        if (MODE == 0) {
          Cf[(size_t)row * N + col] = v;
        } else {
          int b = row >> 11, t = row & (T_DIM - 1);     // M index = b*T + t
          int which = col >> 10, c = col & (C_DIM - 1); // 0:q 1:k 2:v
          int h = c >> 6, d = c & (HS_DIM - 1);
          bf16 bv = (bf16)v;
          size_t bh = (size_t)(b * H_DIM + h);
          if (which == 0)      Qb[(bh * T_DIM + t) * HS_DIM + d] = bv;
          else if (which == 1) Kb[(bh * T_DIM + t) * HS_DIM + d] = bv;
          else                 Vtb[(bh * HS_DIM + d) * T_DIM + t] = bv;  // V transposed
        }
      }
}

// ---------------- flash attention with relative position ----------------
// One 16-row query tile per wave. Q,K: [B,H,T,HS]; Vt: [B,H,HS,T]; Er: [T,HS]; Y: [B,T,C]
__global__ __launch_bounds__(256) void attn_kernel(const bf16* __restrict__ Q,
                                                   const bf16* __restrict__ Kv,
                                                   const bf16* __restrict__ Vt,
                                                   const bf16* __restrict__ Er,
                                                   bf16* __restrict__ Y) {
  __shared__ __align__(16) bf16 sP[8][16 * 32];  // per-wave P staging
  const int lane = threadIdx.x & 31, w = threadIdx.x >> 5;
  const int bh = blockIdx.y, b = bh >> 4, h = bh & 15;
  const int q0 = (blockIdx.x * 8 + w) * 16;
  const int half = lane >> 4, cl = lane & 15;
  const bf16* Qh = Q  + (size_t)bh * T_DIM * HS_DIM;
  const bf16* Kh = Kv + (size_t)bh * T_DIM * HS_DIM;
  const bf16* Vh = Vt + (size_t)bh * HS_DIM * T_DIM;

  bf16x16 aQ[2];
#pragma unroll
  for (int kc = 0; kc < 2; ++kc)
    aQ[kc] = load_frag_a(Qh + (size_t)q0 * HS_DIM + kc * 32, HS_DIM, lane);

  f32x8 O[4] = {};
  float m[8], l[8];
#pragma unroll
  for (int r = 0; r < 8; ++r) { m[r] = -__builtin_inff(); l[r] = 0.f; }
  const float scale = 0.125f;  // 1/sqrt(64)

  for (int k0 = 0; k0 < q0 + 16; k0 += 32) {
    // ---- S = Q @ K^T (two 16x16 column tiles) ----
    f32x8 s[2] = {};
#pragma unroll
    for (int nt = 0; nt < 2; ++nt)
#pragma unroll
      for (int kc = 0; kc < 2; ++kc) {
        bf16x16 bk = load_frag_bt(Kh + (size_t)(k0 + nt * 16) * HS_DIM + kc * 32, HS_DIM, lane);
        s[nt] = wmma_bf16(aQ[kc], bk, s[nt]);
      }

    // ---- G = Q @ ErChunk^T over 48 relative columns; Srel[i,j] = G[i, j-i+15] ----
    const int eb = T_DIM - 16 + k0 - q0;  // Er row for G column 0
    f32x8 g[3] = {};
#pragma unroll
    for (int nt = 0; nt < 3; ++nt) {
      int er = eb + nt * 16 + cl;
      er = er < 0 ? 0 : (er > T_DIM - 1 ? T_DIM - 1 : er);  // clamp (masked anyway)
#pragma unroll
      for (int kc = 0; kc < 2; ++kc) {
        const bf16* p = Er + (size_t)er * HS_DIM + kc * 32 + half * 16;
        bf16x16 be = cat8(*(const bf16x8*)p, *(const bf16x8*)(p + 8));
        g[nt] = wmma_bf16(aQ[kc], be, g[nt]);
      }
    }

    // ---- skew gather + mask + online softmax ----
    float alpha[8];
#pragma unroll
    for (int r = 0; r < 8; ++r) {
      int ii = r + half * 8;
      int ig = q0 + ii;
      int nc0 = cl - ii + 15;        // in [0,30]
      int nc1 = nc0 + 16;            // in [16,46]
      int src0 = (lane & 16) | (nc0 & 15);
      int src1 = (lane & 16) | (nc1 & 15);
      float v0 = __shfl(g[0][r], src0);
      float v1 = __shfl(g[1][r], src0);
      float v2 = __shfl(g[2][r], src0);
      float rel0 = nc0 < 16 ? v0 : (nc0 < 32 ? v1 : v2);
      float u1 = __shfl(g[1][r], src1);
      float u2 = __shfl(g[2][r], src1);
      float rel1 = nc1 < 32 ? u1 : u2;

      int j0 = k0 + cl, j1 = j0 + 16;
      float s0 = (j0 <= ig) ? (s[0][r] + rel0) * scale : -1e30f;
      float s1 = (j1 <= ig) ? (s[1][r] + rel1) * scale : -1e30f;

      float tmax = fmaxf(s0, s1);
      tmax = fmaxf(tmax, __shfl_xor(tmax, 1));
      tmax = fmaxf(tmax, __shfl_xor(tmax, 2));
      tmax = fmaxf(tmax, __shfl_xor(tmax, 4));
      tmax = fmaxf(tmax, __shfl_xor(tmax, 8));
      float mn = fmaxf(m[r], tmax);
      float a  = __expf(m[r] - mn);
      float e0 = __expf(s0 - mn);
      float e1 = __expf(s1 - mn);
      float rs = e0 + e1;
      rs += __shfl_xor(rs, 1);
      rs += __shfl_xor(rs, 2);
      rs += __shfl_xor(rs, 4);
      rs += __shfl_xor(rs, 8);
      l[r] = l[r] * a + rs;
      m[r] = mn;
      alpha[r] = a;
      sP[w][ii * 32 + cl]      = (bf16)e0;
      sP[w][ii * 32 + 16 + cl] = (bf16)e1;
    }

    // rescale running O by alpha (per row)
#pragma unroll
    for (int nt = 0; nt < 4; ++nt)
#pragma unroll
      for (int r = 0; r < 8; ++r)
        O[nt][r] *= alpha[r];

    // per-wave LDS ordering: stores above must land before cross-lane reads below
    asm volatile("s_wait_dscnt 0" ::: "memory");

    bf16x16 aP = load_frag_a(&sP[w][0], 32, lane);
#pragma unroll
    for (int nt = 0; nt < 4; ++nt) {
      bf16x16 bv = load_frag_bt(Vh + (size_t)(nt * 16) * T_DIM + k0, T_DIM, lane);
      O[nt] = wmma_bf16(aP, bv, O[nt]);
    }
    // keep next iteration's sP stores from hoisting above this iteration's reads
    asm volatile("s_wait_dscnt 0" ::: "memory");
  }

#pragma unroll
  for (int r = 0; r < 8; ++r) l[r] = 1.f / l[r];
#pragma unroll
  for (int nt = 0; nt < 4; ++nt)
#pragma unroll
    for (int r = 0; r < 8; ++r) {
      int ii = r + half * 8;
      int t  = q0 + ii;
      int c  = h * HS_DIM + nt * 16 + cl;
      Y[(size_t)(b * T_DIM + t) * C_DIM + c] = (bf16)(O[nt][r] * l[r]);
    }
}

// ---------------- host launcher ----------------
extern "C" void kernel_launch(void* const* d_in, const int* in_sizes, int n_in,
                              void* d_out, int out_size, void* d_ws, size_t ws_size,
                              hipStream_t stream) {
  const float* x  = (const float*)d_in[0];
  const float* Wa = (const float*)d_in[1];
  const float* Wp = (const float*)d_in[2];
  const float* Er = (const float*)d_in[3];

  char* ws = (char*)d_ws;
  size_t off = 0;
  auto take = [&](size_t bytes) -> char* {
    char* p = ws + off;
    off += (bytes + 255) & ~(size_t)255;
    return p;
  };
  const size_t NE = (size_t)B_DIM * T_DIM * C_DIM;  // 4M elems
  bf16* xb  = (bf16*)take(NE * 2);
  bf16* WaT = (bf16*)take((size_t)3 * C_DIM * C_DIM * 2);
  bf16* WpT = (bf16*)take((size_t)C_DIM * C_DIM * 2);
  bf16* Erb = (bf16*)take((size_t)T_DIM * HS_DIM * 2);
  bf16* Qb  = (bf16*)take(NE * 2);
  bf16* Kb  = (bf16*)take(NE * 2);
  bf16* Vtb = (bf16*)take(NE * 2);
  bf16* Yb  = (bf16*)take(NE * 2);

  cvt_f32_bf16<<<2048, 256, 0, stream>>>(x, xb, (int)NE);
  transpose_cvt<<<2048, 256, 0, stream>>>(Wa, WaT, C_DIM, 3 * C_DIM);
  transpose_cvt<<<1024, 256, 0, stream>>>(Wp, WpT, C_DIM, C_DIM);
  cvt_f32_bf16<<<128, 256, 0, stream>>>(Er, Erb, T_DIM * HS_DIM);

  // qkv = x @ W_attn, scattered into Q/K/Vt head layouts
  gemm_wmma<1><<<dim3(32, 24), 256, 0, stream>>>(xb, WaT, nullptr, Qb, Kb, Vtb,
                                                 B_DIM * T_DIM, 3 * C_DIM, C_DIM);
  // attention
  attn_kernel<<<dim3(T_DIM / 128, B_DIM * H_DIM), 256, 0, stream>>>(Qb, Kb, Vtb, Erb, Yb);
  // out = Y @ W_proj (f32)
  gemm_wmma<0><<<dim3(32, 8), 256, 0, stream>>>(Yb, WpT, (float*)d_out, nullptr, nullptr, nullptr,
                                                B_DIM * T_DIM, C_DIM, C_DIM);
}
